// Head_71219147702646
// MI455X (gfx1250) — matrix operand order repaired and checked by reading, not compiled, not run on hardware
//
#include <hip/hip_runtime.h>
#include <hip/hip_bf16.h>

typedef __attribute__((ext_vector_type(16))) _Float16 v16h;
typedef __attribute__((ext_vector_type(8)))  float    v8f;
typedef unsigned int u32;

union F16x {
    v16h v;
    _Float16 h[16];
    u32 u[8];
    uint4 q[2];
};

// A-fragment K index pattern for 16-bit 16x32 A (per CDNA5 ISA 7.12.2):
// VGPR i<4: k = (hi?8:0) + 2*(i&3); i>=4: 16 + (hi?8:0) + 2*(i&3). Halves k,k+1.
__device__ __forceinline__ int a_k(int i, bool hi) {
    return ((i < 4) ? 0 : 16) + (hi ? 8 : 0) + 2 * (i & 3);
}
// Contiguous K index pattern (unused; B fragments here use a_k order): lanes0-15 k=2i, lanes16-31 k=16+2i.
__device__ __forceinline__ int b_k(int i, bool hi) {
    return (hi ? 16 : 0) + 2 * i;
}

__device__ __forceinline__ float redmax16(float t) {
    t = fmaxf(t, __shfl_xor(t, 1));
    t = fmaxf(t, __shfl_xor(t, 2));
    t = fmaxf(t, __shfl_xor(t, 4));
    t = fmaxf(t, __shfl_xor(t, 8));
    return t;
}
__device__ __forceinline__ float redsum16(float t) {
    t += __shfl_xor(t, 1);
    t += __shfl_xor(t, 2);
    t += __shfl_xor(t, 4);
    t += __shfl_xor(t, 8);
    return t;
}

#define BATCH 4
#define TSEQ  4096
#define CDIM  1024
#define HDIM  64

// ---------------------------------------------------------------------------
// Kernel 1: pack Wq/Wk/Wv (fp32 [1024,64]) into f16 B-fragment layout.
// Layout: uint index = ((matrix*32 + kstep)*4 + ntile)*32*8 + lane*8 + i
// so a projection-wave B-frag is lane-contiguous 32B.
// ---------------------------------------------------------------------------
__global__ void k_pack_w(const float* __restrict__ Wq, const float* __restrict__ Wk,
                         const float* __restrict__ Wv, u32* __restrict__ wpack) {
    int t = blockIdx.x * blockDim.x + threadIdx.x;   // 0 .. 98303
    int i    = t & 7;
    int lane = (t >> 3) & 31;
    int nt   = (t >> 8) & 3;
    int ks   = (t >> 10) & 31;
    int m    = t >> 15;                              // 0..2
    const float* W = (m == 0) ? Wq : (m == 1) ? Wk : Wv;
    bool hi = lane >= 16;
    int k   = ks * 32 + a_k(i, hi);
    int col = nt * 16 + (lane & 15);
    union { u32 u; _Float16 h[2]; } p;
    p.h[0] = (_Float16)W[k * HDIM + col];
    p.h[1] = (_Float16)W[(k + 1) * HDIM + col];
    wpack[t] = p.u;
}

// ---------------------------------------------------------------------------
// Kernel 2: fused projections q/k/v = X @ W via WMMA f16 (f32 accumulate).
// One wave computes a 16x64 output tile for one of the three matrices.
// ---------------------------------------------------------------------------
__global__ __launch_bounds__(128) void
k_proj(const float* __restrict__ Xq, const float* __restrict__ Xk,
       const float* __restrict__ Xv, const u32* __restrict__ wpack,
       _Float16* __restrict__ qh, _Float16* __restrict__ kh,
       _Float16* __restrict__ vh) {
    int gw   = blockIdx.x * (blockDim.x >> 5) + (threadIdx.x >> 5); // 0..3071
    int lane = threadIdx.x & 31;
    int m    = gw >> 10;        // which matrix
    int rt   = gw & 1023;       // 16-row tile index over B*T
    const float* X = (m == 0) ? Xq : (m == 1) ? Xk : Xv;
    _Float16*    O = (m == 0) ? qh : (m == 1) ? kh : vh;
    const u32*  wp = wpack + m * 32768;
    bool hi = lane >= 16;
    int  mr = lane & 15;

    v8f acc[4] = {};
    const float* xrow = X + (size_t)(rt * 16 + mr) * CDIM;

    for (int ks = 0; ks < 32; ++ks) {
        F16x a;
#pragma unroll
        for (int i = 0; i < 8; ++i) {
            int k = ks * 32 + a_k(i, hi);
            float2 f = *(const float2*)(xrow + k);
            a.h[2 * i]     = (_Float16)f.x;
            a.h[2 * i + 1] = (_Float16)f.y;
        }
#pragma unroll
        for (int nt = 0; nt < 4; ++nt) {
            F16x b;
            const uint4* bp = (const uint4*)(wp + (((ks * 4 + nt) * 32 + lane) * 8));
            b.q[0] = bp[0];
            b.q[1] = bp[1];
            acc[nt] = __builtin_amdgcn_wmma_f32_16x16x32_f16(
                false, a.v, false, b.v, (short)0, acc[nt], false, false);
        }
    }
    int rbase = rt * 16 + (hi ? 8 : 0);
#pragma unroll
    for (int nt = 0; nt < 4; ++nt) {
        int col = nt * 16 + mr;
#pragma unroll
        for (int r = 0; r < 8; ++r) {
            O[(size_t)(rbase + r) * HDIM + col] = (_Float16)acc[nt][r];
        }
    }
}

// ---------------------------------------------------------------------------
// Kernel 3: repack v (f16 row-major [B*T,64]) into B-fragment layout so each
// V fragment in the attention inner loop is one contiguous 32B load per lane.
// uint index = ((chunk*4 + ntile)*32 + lane)*8 + i, chunk = 32 rows.
// ---------------------------------------------------------------------------
__global__ void k_repack_v(const _Float16* __restrict__ vh, u32* __restrict__ vpack) {
    int t = blockIdx.x * blockDim.x + threadIdx.x;   // 0 .. 524287
    int i    = t & 7;
    int lane = (t >> 3) & 31;
    int nt   = (t >> 8) & 3;
    int c    = t >> 10;                              // chunk 0..511
    bool hi = lane >= 16;
    int k   = c * 32 + a_k(i, hi);
    int col = nt * 16 + (lane & 15);
    union { u32 u; _Float16 h[2]; } p;
    p.h[0] = vh[(size_t)k * HDIM + col];
    p.h[1] = vh[(size_t)(k + 1) * HDIM + col];
    vpack[t] = p.u;
}

// ---------------------------------------------------------------------------
// Kernel 4: causal flash attention. One wave per (batch, 16-query tile).
// Online softmax in fp32; P converted C-layout -> A-layout via per-wave LDS.
// ---------------------------------------------------------------------------
__global__ __launch_bounds__(128) void
k_attn(const _Float16* __restrict__ qh, const _Float16* __restrict__ kh,
       const u32* __restrict__ vpack, float* __restrict__ out) {
    __shared__ _Float16 Pb[4][16 * 34];
    const float c2 = 0.125f * 1.44269504088896340736f;  // H^-0.5 * log2(e)

    int wv   = threadIdx.x >> 5;
    int lane = threadIdx.x & 31;
    int gw   = blockIdx.x * 4 + wv;          // 0..1023
    int b    = gw >> 8;
    int qt   = gw & 255;
    int qs   = qt * 16;
    bool hi  = lane >= 16;
    int  ln  = lane & 15;
    size_t rowbase = (size_t)b * TSEQ;

    // Q fragments (h 0..31 and 32..63), kept in registers for the whole loop.
    F16x aq0, aq1;
    const _Float16* qrow = qh + (rowbase + qs + ln) * HDIM;
#pragma unroll
    for (int i = 0; i < 8; ++i) {
        int k = a_k(i, hi);
        aq0.u[i] = *(const u32*)(qrow + k);
        aq1.u[i] = *(const u32*)(qrow + 32 + k);
    }

    v8f acc[4] = {};
    float mrow[8], lsum[8];
#pragma unroll
    for (int r = 0; r < 8; ++r) { mrow[r] = -1.0e30f; lsum[r] = 0.0f; }

    int qend = qs + 15;
    _Float16* P = Pb[wv];

    for (int j = 0; j <= qend; j += 32) {
        if (j + 32 <= qend) {
            __builtin_prefetch(kh + (rowbase + j + 32 + ln) * HDIM, 0, 1);
        }
        // K fragments: bk[key-half][h-step]; two 16B loads per lane, at h offsets 8*hi and 16+8*hi.
        F16x bk[2][2];
#pragma unroll
        for (int kt = 0; kt < 2; ++kt) {
            const _Float16* krow = kh + (rowbase + j + kt * 16 + ln) * HDIM;
#pragma unroll
            for (int hs = 0; hs < 2; ++hs) {
                const _Float16* p = krow + hs * 32 + (hi ? 8 : 0);
                bk[kt][hs].q[0] = *(const uint4*)(p);
                bk[kt][hs].q[1] = *(const uint4*)(p + 16);
            }
        }
        v8f z = {};
        v8f s0 = __builtin_amdgcn_wmma_f32_16x16x32_f16(false, aq0.v, false, bk[0][0].v, (short)0, z, false, false);
        s0     = __builtin_amdgcn_wmma_f32_16x16x32_f16(false, aq1.v, false, bk[0][1].v, (short)0, s0, false, false);
        v8f s1 = __builtin_amdgcn_wmma_f32_16x16x32_f16(false, aq0.v, false, bk[1][0].v, (short)0, z, false, false);
        s1     = __builtin_amdgcn_wmma_f32_16x16x32_f16(false, aq1.v, false, bk[1][1].v, (short)0, s1, false, false);

        int key0 = j + ln;
        int key1 = key0 + 16;
        float pr0[8], pr1[8];
#pragma unroll
        for (int r = 0; r < 8; ++r) {
            int qg = qs + r + (hi ? 8 : 0);
            float v0 = (key0 <= qg) ? s0[r] * c2 : -1.0e30f;
            float v1 = (key1 <= qg) ? s1[r] * c2 : -1.0e30f;
            float mx = redmax16(fmaxf(v0, v1));
            float mn = fmaxf(mrow[r], mx);
            float al = __builtin_amdgcn_exp2f(mrow[r] - mn);
            float p0 = __builtin_amdgcn_exp2f(v0 - mn);
            float p1 = __builtin_amdgcn_exp2f(v1 - mn);
            lsum[r] = lsum[r] * al + redsum16(p0 + p1);
            mrow[r] = mn;
#pragma unroll
            for (int nt = 0; nt < 4; ++nt) acc[nt][r] = acc[nt][r] * al;
            pr0[r] = p0;
            pr1[r] = p1;
        }
        // C-layout -> A-layout for P via per-wave LDS tile (stride 34 halves).
#pragma unroll
        for (int r = 0; r < 8; ++r) {
            int row = r + (hi ? 8 : 0);
            P[row * 34 + ln]      = (_Float16)pr0[r];
            P[row * 34 + 16 + ln] = (_Float16)pr1[r];
        }
        asm volatile("s_wait_dscnt 0" ::: "memory");
        F16x ap;
#pragma unroll
        for (int i = 0; i < 8; ++i) {
            int k = a_k(i, hi);
            ap.u[i] = *(const u32*)(P + ln * 34 + k);
        }
        // P @ V
        int cidx = (int)((rowbase + j) >> 5);
#pragma unroll
        for (int nt = 0; nt < 4; ++nt) {
            F16x bv;
            const uint4* vp4 = (const uint4*)(vpack + ((size_t)(cidx * 4 + nt) * 32 + lane) * 8);
            bv.q[0] = vp4[0];
            bv.q[1] = vp4[1];
            acc[nt] = __builtin_amdgcn_wmma_f32_16x16x32_f16(
                false, ap.v, false, bv.v, (short)0, acc[nt], false, false);
        }
    }
    // Epilogue: normalize and write fp32 output [B,T,H].
#pragma unroll
    for (int nt = 0; nt < 4; ++nt) {
        int col = nt * 16 + ln;
#pragma unroll
        for (int r = 0; r < 8; ++r) {
            int row = qs + r + (hi ? 8 : 0);
            out[(rowbase + row) * HDIM + col] = acc[nt][r] / lsum[r];
        }
    }
}

extern "C" void kernel_launch(void* const* d_in, const int* in_sizes, int n_in,
                              void* d_out, int out_size, void* d_ws, size_t ws_size,
                              hipStream_t stream) {
    const float* query = (const float*)d_in[0];
    const float* key   = (const float*)d_in[1];
    const float* value = (const float*)d_in[2];
    const float* Wq    = (const float*)d_in[3];
    const float* Wk    = (const float*)d_in[4];
    const float* Wv    = (const float*)d_in[5];

    unsigned char* w = (unsigned char*)d_ws;
    u32*      wpack = (u32*)(w + 0);                 //   393,216 B
    _Float16* qhb   = (_Float16*)(w + 393216);       // 2,097,152 B
    _Float16* khb   = (_Float16*)(w + 2490368);      // 2,097,152 B
    _Float16* vhb   = (_Float16*)(w + 4587520);      // 2,097,152 B
    u32*      vpk   = (u32*)(w + 6684672);           // 2,097,152 B (end 8,781,824)

    k_pack_w<<<384, 256, 0, stream>>>(Wq, Wk, Wv, wpack);
    k_proj<<<768, 128, 0, stream>>>(query, key, value, wpack, qhb, khb, vhb);
    k_repack_v<<<2048, 256, 0, stream>>>(vhb, vpk);
    k_attn<<<256, 128, 0, stream>>>(qhb, khb, vpk, (float*)d_out);
}
